// SupervisedRefiner_86337432584465
// MI455X (gfx1250) — matrix operand order repaired and checked by reading, compiled, not run on hardware
//
#include <hip/hip_runtime.h>
#include <math.h>

typedef __attribute__((ext_vector_type(16))) _Float16 v16h;
typedef __attribute__((ext_vector_type(8)))  _Float16 v8h;
typedef __attribute__((ext_vector_type(8)))  float    v8f;

static constexpr int Bc   = 8;
static constexpr int Nn   = 512;
static constexpr int NSn  = 256;
static constexpr int HID  = 256;
static constexpr int NH   = 8;
static constexpr int DHm  = 32;
static constexpr int SHID = 128;
static constexpr int SDH  = 16;
static constexpr int LATd = 64;

__constant__ int IU_R[21] = {0,0,0,0,0,0, 1,1,1,1,1, 2,2,2,2, 3,3,3, 4,4, 5};
__constant__ int IU_C[21] = {0,1,2,3,4,5, 1,2,3,4,5, 2,3,4,5, 3,4,5, 4,5, 5};

static __device__ __forceinline__ float geluf(float x){ return 0.5f*x*(1.0f+erff(x*0.70710678118654752f)); }
static __device__ __forceinline__ float wave_sum(float v){ for(int m=16;m;m>>=1) v += __shfl_xor(v,m,32); return v; }
static __device__ __forceinline__ float wave_max(float v){ for(int m=16;m;m>>=1) v = fmaxf(v,__shfl_xor(v,m,32)); return v; }

// A-operand fragment: two contiguous 8-float runs at base+0 and base+16 (base = row + k0 + khalf*8).
static __device__ __forceinline__ v16h load_a_frag_fast(const float* __restrict__ base)
{
  const float4 x0 = *(const float4*)(base + 0);
  const float4 x1 = *(const float4*)(base + 4);
  const float4 y0 = *(const float4*)(base + 16);
  const float4 y1 = *(const float4*)(base + 20);
  v16h a;
  a[0]=(_Float16)x0.x; a[1]=(_Float16)x0.y; a[2]=(_Float16)x0.z; a[3]=(_Float16)x0.w;
  a[4]=(_Float16)x1.x; a[5]=(_Float16)x1.y; a[6]=(_Float16)x1.z; a[7]=(_Float16)x1.w;
  a[8]=(_Float16)y0.x; a[9]=(_Float16)y0.y; a[10]=(_Float16)y0.z; a[11]=(_Float16)y0.w;
  a[12]=(_Float16)y1.x; a[13]=(_Float16)y1.y; a[14]=(_Float16)y1.z; a[15]=(_Float16)y1.w;
  return a;
}
// A-operand fragment from 16 consecutive floats (16B-aligned base).
static __device__ __forceinline__ v16h load_contig16(const float* __restrict__ base)
{
  const float4 x0 = *(const float4*)(base + 0);
  const float4 x1 = *(const float4*)(base + 4);
  const float4 x2 = *(const float4*)(base + 8);
  const float4 x3 = *(const float4*)(base + 12);
  v16h r;
  r[0]=(_Float16)x0.x; r[1]=(_Float16)x0.y; r[2]=(_Float16)x0.z; r[3]=(_Float16)x0.w;
  r[4]=(_Float16)x1.x; r[5]=(_Float16)x1.y; r[6]=(_Float16)x1.z; r[7]=(_Float16)x1.w;
  r[8]=(_Float16)x2.x; r[9]=(_Float16)x2.y; r[10]=(_Float16)x2.z; r[11]=(_Float16)x2.w;
  r[12]=(_Float16)x3.x; r[13]=(_Float16)x3.y; r[14]=(_Float16)x3.z; r[15]=(_Float16)x3.w;
  return r;
}

// ---------------- generic WMMA GEMM: C = beta*C + act(A[M,K] @ W[K,N] + bias) ----------------
// Block = 4 waves = 64 rows x 64 cols. Each wave: 1 A-frag x 4 B-frags (LDS) -> 4 wmma per K-step.
// FULL=1: K%32==0 -> no K guards; row/col indices clamped for loads (stores stay guarded).
template<int FULL>
__global__ void gemm_kernel(const float* __restrict__ A, const float* __restrict__ W,
                            const float* __restrict__ bias, float* __restrict__ C,
                            int M, int N, int K, int act, int beta)
{
  __shared__ _Float16 wtile[2 * 4 * 512];  // [buf][ct][lane][j]
  const int tid   = threadIdx.x;
  const int lane  = tid & 31;
  const int wid   = tid >> 5;
  const int khalf = lane >> 4;
  const int ncol0 = blockIdx.x * 64;
  const int rowA  = blockIdx.y * 64 + wid * 16 + (lane & 15);
  const int wcol  = tid & 15;      // staging: column within 16-tile
  const int wkrow = tid >> 4;      // staging: base k row (0..7)
  const bool rowOK = rowA < M;
  const unsigned rowL = (unsigned)(rowOK ? rowA : (M - 1));
  const unsigned uN = (unsigned)N, uK = (unsigned)K;
  unsigned wcolC[4];
#pragma unroll
  for (int ct = 0; ct < 4; ++ct) {
    int c = ncol0 + ct * 16 + wcol;
    wcolC[ct] = (unsigned)((c < N) ? c : (N - 1));
  }
  v8f acc[4];
  { v8f z = {};
#pragma unroll
    for (int ct = 0; ct < 4; ++ct) acc[ct] = z; }
  int buf = 0;
  for (int k0 = 0; k0 < K; k0 += 32, buf ^= 1) {
    _Float16* wt = wtile + buf * 2048;
    if (FULL) {
#pragma unroll
      for (int ct = 0; ct < 4; ++ct) {
        const float v0 = W[(unsigned)(k0 + wkrow +  0) * uN + wcolC[ct]];
        const float v1 = W[(unsigned)(k0 + wkrow +  8) * uN + wcolC[ct]];
        const float v2 = W[(unsigned)(k0 + wkrow + 16) * uN + wcolC[ct]];
        const float v3 = W[(unsigned)(k0 + wkrow + 24) * uN + wcolC[ct]];
        _Float16* wtc = wt + ct * 512;
        const int k_0 = wkrow, k_1 = wkrow + 8, k_2 = wkrow + 16, k_3 = wkrow + 24;
        wtc[(((k_0 >> 4) << 4) + wcol) * 16 + (k_0 & 15)] = (_Float16)v0;
        wtc[(((k_1 >> 4) << 4) + wcol) * 16 + (k_1 & 15)] = (_Float16)v1;
        wtc[(((k_2 >> 4) << 4) + wcol) * 16 + (k_2 & 15)] = (_Float16)v2;
        wtc[(((k_3 >> 4) << 4) + wcol) * 16 + (k_3 & 15)] = (_Float16)v3;
      }
      // prefetch next K-chunk of W (clamped; lowers to global_prefetch_b8)
      __builtin_prefetch(W + (unsigned)(((k0 + 32 < K) ? (k0 + 32) : 0) + wkrow) * uN + wcolC[0], 0, 1);
    } else {
#pragma unroll
      for (int ct = 0; ct < 4; ++ct) {
        _Float16* wtc = wt + ct * 512;
#pragma unroll
        for (int p = 0; p < 4; ++p) {
          int k = p * 8 + wkrow;
          float v = 0.0f;
          if (ncol0 + ct * 16 + wcol < N && k0 + k < K)
            v = W[(unsigned)(k0 + k) * uN + (unsigned)(ncol0 + ct * 16 + wcol)];
          wtc[(((k >> 4) << 4) + wcol) * 16 + (k & 15)] = (_Float16)v;
        }
      }
    }
    __syncthreads();
    v16h a;
    if (FULL) {
      a = load_contig16(A + rowL * uK + (unsigned)(k0 + (khalf << 4)));
    } else {
#pragma unroll
      for (int j = 0; j < 16; ++j) {
        int ka = k0 + (khalf << 4) + j;
        a[j] = (_Float16)((rowOK && ka < K) ? A[rowL * uK + (unsigned)ka] : 0.0f);
      }
    }
#pragma unroll
    for (int ct = 0; ct < 4; ++ct) {
      v16h b = *(const v16h*)(wt + ct * 512 + lane * 16);
      acc[ct] = __builtin_amdgcn_wmma_f32_16x16x32_f16(false, a, false, b, (short)0, acc[ct], false, false);
    }
  }
  const int  mbase = blockIdx.y * 64 + wid * 16 + (khalf << 3);
  const bool fullM = (blockIdx.y * 64 + wid * 16 + 15) < M;   // wave-uniform
#pragma unroll
  for (int ct = 0; ct < 4; ++ct) {
    const int col = ncol0 + ct * 16 + (lane & 15);
    const bool colOK = (col < N);
    const float bi = (colOK && bias) ? bias[col] : 0.0f;
    if (colOK && fullM) {
      float* cp = C + (unsigned)mbase * uN + (unsigned)col;
      if (beta) {
#pragma unroll
        for (int r = 0; r < 8; ++r) {
          float v = acc[ct][r] + bi;
          if (act == 1) v = geluf(v);
          cp[(unsigned)r * uN] += v;
        }
      } else {
#pragma unroll
        for (int r = 0; r < 8; ++r) {
          float v = acc[ct][r] + bi;
          if (act == 1) v = geluf(v);
          cp[(unsigned)r * uN] = v;
        }
      }
    } else if (colOK) {
#pragma unroll
      for (int r = 0; r < 8; ++r) {
        int m = mbase + r;
        if (m < M) {
          float v = acc[ct][r] + bi;
          if (act == 1) v = geluf(v);
          unsigned idx = (unsigned)m * uN + (unsigned)col;
          C[idx] = (beta ? C[idx] : 0.0f) + v;
        }
      }
    }
  }
}

// ---------------- LayerNorm (one wave per row, Y may alias X) ----------------
__global__ void ln_kernel(const float* X, const float* g, const float* bb, float* Y, int D)
{
  const unsigned row = blockIdx.x;
  const int lane = threadIdx.x;
  const float* xr = X + row * (unsigned)D;
  float s = 0.f, s2 = 0.f;
  for (int i = lane; i < D; i += 32) { float v = xr[i]; s += v; s2 += v*v; }
  s = wave_sum(s); s2 = wave_sum(s2);
  float mean = s / (float)D;
  float inv = rsqrtf(fmaxf(s2/(float)D - mean*mean, 0.0f) + 1e-5f);
  float* yr = Y + row * (unsigned)D;
  for (int i = lane; i < D; i += 32) yr[i] = (xr[i] - mean) * inv * g[i] + bb[i];
}

__global__ void add_role_cond_kernel(float* h, const int* __restrict__ roles,
                                     const float* __restrict__ emb, const float* __restrict__ cond,
                                     int rows, int Nper, int D)
{
  int idx = blockIdx.x * blockDim.x + threadIdx.x;
  if (idx >= rows * D) return;
  int row = idx / D, d = idx - row * D;
  float v = h[idx] + emb[roles[row] * D + d];
  if (cond) v += cond[(row / Nper) * D + d];
  h[idx] = v;
}

__global__ void mechfeat_kernel(const float* T, const float* Cs, float* F)
{
  int idx = blockIdx.x * blockDim.x + threadIdx.x;
  if (idx >= Bc * 63) return;
  int b = idx / 63, i = idx % 63;
  float v;
  if (i < 36) v = T[b*36 + i];
  else if (i < 57) { int k = i - 36; v = Cs[b*36 + IU_R[k]*6 + IU_C[k]]; }
  else { int d = i - 57; v = T[b*36 + d*7] - Cs[b*36 + d*7]; }
  F[idx] = v;
}

__global__ void sinus_kernel(const float* t, float* S)
{
  int idx = blockIdx.x * blockDim.x + threadIdx.x;
  if (idx >= Bc * 128) return;
  int b = idx / 128, i = idx % 128;
  float f = expf(-6.907755278982137f * (float)i / 127.0f);  // 1000^{-i/127}
  float a = t[b] * f;
  S[b*256 + i] = sinf(a);
  S[b*256 + 128 + i] = cosf(a);
}

__global__ void noise_kernel(const float* ft, const float* pn, const float* an, float* O)
{
  int b = threadIdx.x;
  if (b < Bc) { O[b*3] = ft[b]; O[b*3+1] = pn[b]; O[b*3+2] = an[b]; }
}

__global__ void build_ks_kernel(const float* X, const float* T, float* KS, int Nq, int D)
{
  int idx = blockIdx.x * blockDim.x + threadIdx.x;
  int total = Bc * (Nq + 1) * D;
  if (idx >= total) return;
  int d = idx % D; int row = idx / D;
  int n = row % (Nq + 1); int b = row / (Nq + 1);
  KS[idx] = (n < Nq) ? X[(unsigned)(b * Nq + n) * (unsigned)D + (unsigned)d] : T[b * D + d];
}

__global__ void meanpool_kernel(const float* X, float* O, int Nrows, int D)
{
  int idx = blockIdx.x * blockDim.x + threadIdx.x;
  if (idx >= Bc * D) return;
  int b = idx / D, d = idx % D;
  float s = 0.f;
  for (int n = 0; n < Nrows; ++n) s += X[(unsigned)(b * Nrows + n) * (unsigned)D + (unsigned)d];
  O[idx] = s / (float)Nrows;
}

// ---------------- attention logits ----------------
// One wave owns 64 query rows (4 n-tiles; Q-frags live in regs) and sweeps all m-tiles:
// Q read once; each K-frag feeds 4 back-to-back wmma; one store guard per m-tile.
__global__ void attn_logits_kernel(const float* __restrict__ Q, const float* __restrict__ Kb,
                                   float* __restrict__ L, int H, int Nq, int Mk, int Ms,
                                   int D, int dh, float scale)
{
  const int lane = threadIdx.x; const int khalf = lane >> 4;
  const int nt0 = blockIdx.y * 4;
  const int b = blockIdx.z / H, h = blockIdx.z % H;
  const unsigned uD = (unsigned)D, uMs = (unsigned)Ms;
  v16h qa[4];
#pragma unroll
  for (int qt = 0; qt < 4; ++qt) {
    const int n = (nt0 + qt) * 16 + (lane & 15);       // always < Nq
    const float* qb = Q + (unsigned)(b*Nq + n)*uD + (unsigned)(h*dh + (khalf << 3));
    const float4 x0 = *(const float4*)(qb + 0);
    const float4 x1 = *(const float4*)(qb + 4);
    qa[qt][0]=(_Float16)x0.x; qa[qt][1]=(_Float16)x0.y; qa[qt][2]=(_Float16)x0.z; qa[qt][3]=(_Float16)x0.w;
    qa[qt][4]=(_Float16)x1.x; qa[qt][5]=(_Float16)x1.y; qa[qt][6]=(_Float16)x1.z; qa[qt][7]=(_Float16)x1.w;
    if (dh == 32) {                                    // uniform branch
      const float4 y0 = *(const float4*)(qb + 16);
      const float4 y1 = *(const float4*)(qb + 20);
      qa[qt][8]=(_Float16)y0.x; qa[qt][9]=(_Float16)y0.y; qa[qt][10]=(_Float16)y0.z; qa[qt][11]=(_Float16)y0.w;
      qa[qt][12]=(_Float16)y1.x; qa[qt][13]=(_Float16)y1.y; qa[qt][14]=(_Float16)y1.z; qa[qt][15]=(_Float16)y1.w;
    } else {
#pragma unroll
      for (int j = 8; j < 16; ++j) qa[qt][j] = (_Float16)0.0f;
    }
  }
  const int boff = (khalf << 3);
  const int Mtiles = (Mk + 15) >> 4;
  for (int mt = 0; mt < Mtiles; ++mt) {
    const int mc = mt * 16 + (lane & 15);
    const int mL = (mc < Mk) ? mc : (Mk - 1);
    v16h bf = load_a_frag_fast(Kb + (unsigned)(b*Mk + mL)*uD + (unsigned)(h*dh + boff));
    if (dh == 16) {
#pragma unroll
      for (int j = 8; j < 16; ++j) bf[j] = (_Float16)0.0f;
    }
    // 4 back-to-back wmma (hazard windows overlap), then one guarded store clause
    v8f acc[4];
#pragma unroll
    for (int qt = 0; qt < 4; ++qt) {
      v8f z = {};
      acc[qt] = __builtin_amdgcn_wmma_f32_16x16x32_f16(false, qa[qt], false, bf, (short)0, z, false, false);
    }
    if (mc < Mk) {
      float* lp0 = L + (unsigned)((b*H + h)*Nq + nt0*16 + (khalf << 3)) * uMs + (unsigned)mc;
#pragma unroll
      for (int qt = 0; qt < 4; ++qt) {
        float* lp = lp0 + (unsigned)(qt * 16) * uMs;
#pragma unroll
        for (int r = 0; r < 8; ++r)
          lp[(unsigned)r * uMs] = acc[qt][r] * scale;
      }
    }
  }
}

// ---------------- bias + softmax + cond reweight (mode: 0 free, 1 distance, 2 conn, 3 stress) ----
__global__ void attn_softmax_kernel(float* L, const float* __restrict__ pos,
                                    const float* __restrict__ adj, const float* __restrict__ ehc,
                                    int H, int Nq, int Mk, int Ms, int mode)
{
  const int rowid = blockIdx.x;
  const int lane = threadIdx.x;
  const int n = rowid % Nq;
  const int h = (rowid / Nq) % H;
  const int b = rowid / (Nq * H);
  float* Lr = L + (unsigned)rowid * (unsigned)Ms;
  float px = 0.f, py = 0.f, pz = 0.f;
  if (mode == 1) { const float* pp = pos + (unsigned)(b*Nq + n)*3u; px = pp[0]; py = pp[1]; pz = pp[2]; }
  const float kls = 10.330578512396694f; // 1/(2*0.22^2)
  float mx = -3.0e38f;
  for (int m = lane; m < Mk; m += 32) {
    float v = Lr[m];
    if (m < Nq) {
      if (mode == 1) {
        const float* pq = pos + (unsigned)(b*Nq + m)*3u;
        float dx = px-pq[0], dy = py-pq[1], dz = pz-pq[2];
        float c = expf(-(dx*dx + dy*dy + dz*dz) * kls);
        v += 2.0f*c - 1.0f;
      } else if (mode == 2) {
        v += 2.0f*adj[(unsigned)(b*Nq + n)*(unsigned)Nq + (unsigned)m] - 1.0f;
      } else if (mode == 3) {
        v += ehc[(unsigned)((b*H + h)*Nq + n)*(unsigned)Nq + (unsigned)m];
      }
    }
    Lr[m] = v;
    mx = fmaxf(mx, v);
  }
  mx = wave_max(mx);
  float s = 0.f;
  for (int m = lane; m < Mk; m += 32) { float e = expf(Lr[m]-mx); Lr[m] = e; s += e; }
  s = wave_sum(s);
  float inv = 1.0f / s;
  if (mode == 1 || mode == 2) {
    float s2 = 0.f;
    for (int m = lane; m < Mk; m += 32) {
      float sc = 1.0f;
      if (m < Nq) {
        float c;
        if (mode == 1) {
          const float* pq = pos + (unsigned)(b*Nq + m)*3u;
          float dx = px-pq[0], dy = py-pq[1], dz = pz-pq[2];
          c = expf(-(dx*dx + dy*dy + dz*dz) * kls);
        } else c = adj[(unsigned)(b*Nq + n)*(unsigned)Nq + (unsigned)m];
        sc = 0.25f + 0.75f*c;
      }
      float w = Lr[m] * inv * sc;
      Lr[m] = w; s2 += w;
    }
    s2 = wave_sum(s2);
    float inv2 = 1.0f / fmaxf(s2, 1e-6f);
    for (int m = lane; m < Mk; m += 32) Lr[m] *= inv2;
  } else {
    for (int m = lane; m < Mk; m += 32) Lr[m] *= inv;
  }
}

// ---------------- att = W @ V  ->  O[b,n, h*dh + d] ----------------
// One wave owns 64 query rows (4 n-tiles) x NDT d-tiles; V-frag feeds 4 wmma per chunk.
template<int NDT>
__global__ void attn_av_kernel(const float* __restrict__ Wt, const float* __restrict__ V,
                               float* __restrict__ O, int H, int Nq, int Mk, int Ms, int D)
{
  const int dh = NDT * 16;
  const int lane = threadIdx.x; const int khalf = lane >> 4;
  const int nt0 = blockIdx.y * 4;
  const int b = blockIdx.z / H, h = blockIdx.z % H;
  const unsigned uD = (unsigned)D, uMs = (unsigned)Ms;
  const float* wrow0 = Wt + (unsigned)((b*H + h)*Nq + nt0*16 + (lane & 15)) * uMs;
  v8f acc[4][NDT];
  { v8f z = {};
#pragma unroll
    for (int qt = 0; qt < 4; ++qt)
#pragma unroll
      for (int dt = 0; dt < NDT; ++dt) acc[qt][dt] = z; }
  const int Kfull = Mk & ~31;
  for (int k0 = 0; k0 < Kfull; k0 += 32) {       // unguarded full chunks
    v16h af[4];
#pragma unroll
    for (int qt = 0; qt < 4; ++qt)
      af[qt] = load_a_frag_fast(wrow0 + (unsigned)(qt * 16) * uMs + (unsigned)(k0 + (khalf << 3)));
#pragma unroll
    for (int dt = 0; dt < NDT; ++dt) {
      const float* vcol = V + (unsigned)(b*Mk)*uD + (unsigned)(h*dh + dt*16 + (lane & 15));
      v16h bf;
#pragma unroll
      for (int j = 0; j < 16; ++j)
        bf[j] = (_Float16)vcol[(unsigned)(k0 + ((j >> 3) << 4) + (khalf << 3) + (j & 7)) * uD];
#pragma unroll
      for (int qt = 0; qt < 4; ++qt)
        acc[qt][dt] = __builtin_amdgcn_wmma_f32_16x16x32_f16(false, af[qt], false, bf, (short)0, acc[qt][dt], false, false);
    }
  }
  if (Kfull < Mk) {                               // guarded tail chunk
    const int k0 = Kfull;
    v16h af[4];
#pragma unroll
    for (int qt = 0; qt < 4; ++qt) {
#pragma unroll
      for (int j = 0; j < 16; ++j) {
        int ka = k0 + ((j >> 3) << 4) + (khalf << 3) + (j & 7);
        af[qt][j] = (_Float16)((ka < Mk) ? wrow0[(unsigned)(qt * 16) * uMs + (unsigned)ka] : 0.0f);
      }
    }
#pragma unroll
    for (int dt = 0; dt < NDT; ++dt) {
      const float* vcol = V + (unsigned)(b*Mk)*uD + (unsigned)(h*dh + dt*16 + (lane & 15));
      v16h bf;
#pragma unroll
      for (int j = 0; j < 16; ++j) {
        int kb = k0 + ((j >> 3) << 4) + (khalf << 3) + (j & 7);
        bf[j] = (_Float16)((kb < Mk) ? vcol[(unsigned)kb * uD] : 0.0f);
      }
#pragma unroll
      for (int qt = 0; qt < 4; ++qt)
        acc[qt][dt] = __builtin_amdgcn_wmma_f32_16x16x32_f16(false, af[qt], false, bf, (short)0, acc[qt][dt], false, false);
    }
  }
#pragma unroll
  for (int qt = 0; qt < 4; ++qt)
#pragma unroll
    for (int dt = 0; dt < NDT; ++dt) {
      float* op = O + (unsigned)(b*Nq + (nt0 + qt)*16 + (khalf << 3))*uD + (unsigned)(h*dh + dt*16 + (lane & 15));
#pragma unroll
      for (int r = 0; r < 8; ++r)
        op[(unsigned)r * uD] = acc[qt][dt][r];
    }
}

// ---------------- fused per-edge MLP: evm[e,6] -> gelu(256) -> 8, into EHC[b,h,i,j] ----------------
// Grid covers all B*N*N edges exactly -> no range guards needed.
__global__ void edge_mlp_kernel(const float* __restrict__ EVM, const float* __restrict__ W1,
                                const float* __restrict__ B1, const float* __restrict__ W2,
                                const float* __restrict__ B2, float* __restrict__ EHC)
{
  __shared__ _Float16 hid[4][16 * HID];
  const int lane  = threadIdx.x & 31;
  const int wid   = threadIdx.x >> 5;
  const int khalf = lane >> 4;
  const unsigned e0 = ((unsigned)blockIdx.x * 4 + wid) * 16;
  const bool live = (khalf == 0);                 // only K=0..5 exists
  v16h a;
  {
    const unsigned e = e0 + (lane & 15);
    const float2* p = (const float2*)(EVM + (size_t)e * 6);   // 8B-aligned (e*24B)
    float2 q0 = p[0], q1 = p[1], q2 = p[2];
    a[0] = (_Float16)(live ? q0.x : 0.0f);
    a[1] = (_Float16)(live ? q0.y : 0.0f);
    a[2] = (_Float16)(live ? q1.x : 0.0f);
    a[3] = (_Float16)(live ? q1.y : 0.0f);
    a[4] = (_Float16)(live ? q2.x : 0.0f);
    a[5] = (_Float16)(live ? q2.y : 0.0f);
#pragma unroll
    for (int j = 6; j < 16; ++j) a[j] = (_Float16)0.0f;
  }
  for (int ct = 0; ct < HID / 16; ++ct) {
    const int c = ct * 16 + (lane & 15);
    v16h bf;
#pragma unroll
    for (int j = 0; j < 6; ++j) {
      float v = W1[(unsigned)j * (unsigned)HID + (unsigned)c];   // unguarded; masked below
      bf[j] = (_Float16)(live ? v : 0.0f);
    }
#pragma unroll
    for (int j = 6; j < 16; ++j) bf[j] = (_Float16)0.0f;
    v8f acc = {};
    acc = __builtin_amdgcn_wmma_f32_16x16x32_f16(false, a, false, bf, (short)0, acc, false, false);
#pragma unroll
    for (int r = 0; r < 8; ++r) {
      int m = r + (khalf << 3);
      hid[wid][m * HID + c] = (_Float16)geluf(acc[r] + B1[c]);
    }
  }
  __syncthreads();
  const int hc = lane & 15;
  const _Float16* hrow = &hid[wid][(lane & 15) * HID];
  v8f acc2 = {};
  for (int k0 = 0; k0 < HID; k0 += 32) {
    v16h a2;
    v8h lo = *(const v8h*)(hrow + k0 + (khalf << 3));
    v8h hi = *(const v8h*)(hrow + k0 + 16 + (khalf << 3));
#pragma unroll
    for (int j = 0; j < 8; ++j) { a2[j] = lo[j]; a2[8 + j] = hi[j]; }
    v16h bf;
#pragma unroll
    for (int j = 0; j < 16; ++j) {
      float v = W2[(unsigned)(k0 + ((j >> 3) << 4) + (khalf << 3) + (j & 7)) * (unsigned)NH + (unsigned)(hc & 7)];
      bf[j] = (_Float16)((hc < NH) ? v : 0.0f);
    }
    acc2 = __builtin_amdgcn_wmma_f32_16x16x32_f16(false, a2, false, bf, (short)0, acc2, false, false);
  }
  if (hc < NH) {
#pragma unroll
    for (int r = 0; r < 8; ++r) {
      unsigned e = e0 + r + (khalf << 3);
      unsigned bidx = e >> 18;                    // / (512*512)
      unsigned rem  = e & 0x3FFFFu;
      unsigned i = rem >> 9, j2 = rem & 511u;
      EHC[(((bidx*NH + (unsigned)hc)*(unsigned)Nn + i)*(unsigned)Nn + j2)] = acc2[r] + B2[hc];
    }
  }
}

// ---------------- pred_adj = sigmoid(lat.lat^T/8), adj_vel = pred_adj - adj ----------------
__global__ void adj_head_kernel(const float* __restrict__ LATp, const float* __restrict__ ADJ,
                                float* __restrict__ adjv, float* __restrict__ padj)
{
  const int lane = threadIdx.x; const int khalf = lane >> 4;
  const int mt = blockIdx.x, nt = blockIdx.y, b = blockIdx.z;
  const int n  = nt * 16 + (lane & 15);
  const int mc = mt * 16 + (lane & 15);
  v8f acc = {};
#pragma unroll
  for (int k0 = 0; k0 < LATd; k0 += 32) {
    v16h a  = load_a_frag_fast(LATp + (unsigned)(b*Nn + n)*(unsigned)LATd + (unsigned)(k0 + (khalf << 3)));
    v16h bf = load_a_frag_fast(LATp + (unsigned)(b*Nn + mc)*(unsigned)LATd + (unsigned)(k0 + (khalf << 3)));
    acc = __builtin_amdgcn_wmma_f32_16x16x32_f16(false, a, false, bf, (short)0, acc, false, false);
  }
  const unsigned base = (unsigned)(b*Nn + nt*16 + (khalf << 3))*(unsigned)Nn + (unsigned)mc;
#pragma unroll
  for (int r = 0; r < 8; ++r) {
    float p = 1.0f / (1.0f + expf(-acc[r] * 0.125f));  // /sqrt(64)
    unsigned idx = base + (unsigned)r * (unsigned)Nn;
    padj[idx] = p;
    adjv[idx] = p - ADJ[idx];
  }
}

// ================= host =================
struct LinP { const float* b; const float* w; };
struct MlpP { LinP l1, l2; };
struct LNP  { const float* b; const float* g; };
struct BlockP { MlpP ff; LinP k; LNP n1, n2; LinP o, q, v; };

// workspace offsets (floats); attention weight rows padded to stride 520
static constexpr int MSmax = 520;
static constexpr size_t SZ_H   = (size_t)Bc*Nn*HID;
static constexpr size_t SZ_KS  = (size_t)Bc*(Nn+1)*HID;
static constexpr size_t OFF_H   = 0;
static constexpr size_t OFF_X   = OFF_H + SZ_H;
static constexpr size_t OFF_KS  = OFF_X + SZ_H;
static constexpr size_t OFF_Q   = OFF_KS + SZ_KS;
static constexpr size_t OFF_K   = OFF_Q + SZ_H;
static constexpr size_t OFF_V   = OFF_K + SZ_KS;
static constexpr size_t OFF_W   = OFF_V + SZ_KS;
static constexpr size_t SZ_W    = (size_t)Bc*NH*Nn*MSmax;
static constexpr size_t OFF_ATT = OFF_W + SZ_W;
static constexpr size_t OFF_FFH = OFF_ATT + SZ_H;
static constexpr size_t SZ_FFH  = (size_t)Bc*Nn*4*HID;
static constexpr size_t OFF_EHC = OFF_FFH + SZ_FFH;
static constexpr size_t SZ_EHC  = (size_t)Bc*NH*Nn*Nn;
static constexpr size_t OFF_HS  = OFF_EHC + SZ_EHC;
static constexpr size_t SZ_HS   = (size_t)Bc*NSn*SHID;
static constexpr size_t OFF_COND = OFF_HS + SZ_HS;
static constexpr size_t OFF_TOK  = OFF_COND + (size_t)Bc*HID;
static constexpr size_t OFF_TMPB = OFF_TOK + (size_t)Bc*HID;
static constexpr size_t OFF_TOKIN= OFF_TMPB + (size_t)Bc*HID;
static constexpr size_t OFF_FEAT = OFF_TOKIN + (size_t)Bc*SHID;
static constexpr size_t OFF_SIN  = OFF_FEAT + (size_t)Bc*64;
static constexpr size_t OFF_NSE  = OFF_SIN + (size_t)Bc*HID;

static constexpr size_t OUT_POS  = 0;
static constexpr size_t OUT_ADJV = (size_t)Bc*Nn*3;
static constexpr size_t OUT_PADJ = OUT_ADJV + (size_t)Bc*Nn*Nn;
static constexpr size_t OUT_LAT  = OUT_PADJ + (size_t)Bc*Nn*Nn;

extern "C" void kernel_launch(void* const* d_in, const int* in_sizes, int n_in,
                              void* d_out, int out_size, void* d_ws, size_t ws_size,
                              hipStream_t stream)
{
  (void)in_sizes; (void)n_in; (void)out_size; (void)ws_size;
  const float* positions = (const float*)d_in[0];
  const int*   roles     = (const int*)d_in[1];
  const float* adjacency = (const float*)d_in[2];
  const float* disp      = (const float*)d_in[3];
  const float* evm       = (const float*)d_in[4];
  const float* tstiff    = (const float*)d_in[5];
  const float* cstiff    = (const float*)d_in[6];
  const float* ftimes    = (const float*)d_in[7];
  const float* pnoise    = (const float*)d_in[8];
  const float* anoise    = (const float*)d_in[9];
  const float* spos      = (const float*)d_in[10];
  const int*   sroles    = (const int*)d_in[11];
  const float* sadj      = (const float*)d_in[12];

  // ---- params: jax pytree flatten order (dict keys sorted; "b"<"w", "b"<"g") ----
  int cur = 13;
  auto nxt = [&]() { return (const float*)d_in[cur++]; };
  auto getLin = [&]() { LinP L; L.b = nxt(); L.w = nxt(); return L; };
  auto getMlp = [&]() { MlpP m; m.l1 = getLin(); m.l2 = getLin(); return m; };
  auto getLN  = [&]() { LNP l; l.b = nxt(); l.g = nxt(); return l; };
  auto getBlk = [&]() { BlockP bl; bl.ff = getMlp(); bl.k = getLin(); bl.n1 = getLN();
                        bl.n2 = getLN(); bl.o = getLin(); bl.q = getLin(); bl.v = getLin(); return bl; };

  MlpP dispmlp = getMlp();
  MlpP edgem   = getMlp();
  LNP  fnorm   = getLN();
  LNP  innorm  = getLN();
  MlpP lathead = getMlp();
  BlockP L[6]; for (int i = 0; i < 6; ++i) L[i] = getBlk();
  MlpP mech    = getMlp();
  MlpP noisem  = getMlp();
  MlpP poshead = getMlp();
  MlpP posmlp  = getMlp();
  const float* role_emb = nxt();
  LNP  s_final = getLN();
  LNP  s_in    = getLN();
  BlockP SL[3]; for (int i = 0; i < 3; ++i) SL[i] = getBlk();
  MlpP sposmlp = getMlp();
  const float* srole_emb = nxt();
  MlpP tokproj = getMlp();
  MlpP timem   = getMlp();

  float* ws  = (float*)d_ws;
  float* out = (float*)d_out;

  auto gemm = [&](const float* A, LinP Lp, float* C, int M, int N, int K, int act, int beta) {
    dim3 grid((N + 63) / 64, (M + 63) / 64);
    if ((K & 31) == 0)
      gemm_kernel<1><<<grid, 128, 0, stream>>>(A, Lp.w, Lp.b, C, M, N, K, act, beta);
    else
      gemm_kernel<0><<<grid, 128, 0, stream>>>(A, Lp.w, Lp.b, C, M, N, K, act, beta);
  };
  auto mlp2 = [&](const float* A, MlpP m, float* C, int M, int K, int Hd, int N, int beta) {
    gemm(A, m.l1, ws + OFF_FFH, M, Hd, K, 1, 0);
    gemm(ws + OFF_FFH, m.l2, C, M, N, Hd, 0, beta);
  };

  // ---- cond [B,HID] ----
  mechfeat_kernel<<<2, 256, 0, stream>>>(tstiff, cstiff, ws + OFF_FEAT);
  mlp2(ws + OFF_FEAT, mech, ws + OFF_COND, Bc, 63, HID, HID, 0);
  sinus_kernel<<<4, 256, 0, stream>>>(ftimes, ws + OFF_SIN);
  mlp2(ws + OFF_SIN, timem, ws + OFF_COND, Bc, HID, HID, HID, 1);
  noise_kernel<<<1, 32, 0, stream>>>(ftimes, pnoise, anoise, ws + OFF_NSE);
  mlp2(ws + OFF_NSE, noisem, ws + OFF_COND, Bc, 3, HID, HID, 1);

  // ---- node embed ----
  mlp2(positions, posmlp, ws + OFF_H, Bc*Nn, 3, HID, HID, 0);
  mlp2(disp, dispmlp, ws + OFF_H, Bc*Nn, 18, HID, HID, 1);
  add_role_cond_kernel<<<(Bc*Nn*HID + 255) / 256, 256, 0, stream>>>(
      ws + OFF_H, roles, role_emb, ws + OFF_COND, Bc*Nn, Nn, HID);
  ln_kernel<<<Bc*Nn, 32, 0, stream>>>(ws + OFF_H, innorm.g, innorm.b, ws + OFF_H, HID);

  // ---- fused per-edge stress MLP -> EHC[B,H,N,N] ----
  edge_mlp_kernel<<<(int)(((size_t)Bc*Nn*Nn / 64)), 128, 0, stream>>>(
      evm, edgem.l1.w, edgem.l1.b, edgem.l2.w, edgem.l2.b, ws + OFF_EHC);

  // ---- generic attention block ----
  auto attn_layer = [&](const BlockP& bp, float* hbuf, int Nq, int D, int dh,
                        int mode, const float* posp, const float* adjp, const float* ehcp,
                        const float* ctx) {
    const int Mk = Nq + (ctx ? 1 : 0);
    const int Ms = (Mk + 7) & ~7;
    ln_kernel<<<Bc*Nq, 32, 0, stream>>>(hbuf, bp.n1.g, bp.n1.b, ws + OFF_X, D);
    const float* Ksrc = ws + OFF_X;
    if (ctx) {
      ln_kernel<<<Bc, 32, 0, stream>>>(ctx, bp.n1.g, bp.n1.b, ws + OFF_TMPB, D);
      int tot = Bc * Mk * D;
      build_ks_kernel<<<(tot + 255) / 256, 256, 0, stream>>>(ws + OFF_X, ws + OFF_TMPB, ws + OFF_KS, Nq, D);
      Ksrc = ws + OFF_KS;
    }
    gemm(ws + OFF_X, bp.q, ws + OFF_Q, Bc*Nq, D, D, 0, 0);
    gemm(Ksrc, bp.k, ws + OFF_K, Bc*Mk, D, D, 0, 0);
    gemm(Ksrc, bp.v, ws + OFF_V, Bc*Mk, D, D, 0, 0);
    dim3 gl(1, Nq / 64, Bc*NH);
    attn_logits_kernel<<<gl, 32, 0, stream>>>(ws + OFF_Q, ws + OFF_K, ws + OFF_W,
                                              NH, Nq, Mk, Ms, D, dh, 1.0f / sqrtf((float)dh));
    attn_softmax_kernel<<<Bc*NH*Nq, 32, 0, stream>>>(ws + OFF_W, posp, adjp, ehcp, NH, Nq, Mk, Ms, mode);
    dim3 ga(1, Nq / 64, Bc*NH);
    if (dh == 32)
      attn_av_kernel<2><<<ga, 32, 0, stream>>>(ws + OFF_W, ws + OFF_V, ws + OFF_ATT, NH, Nq, Mk, Ms, D);
    else
      attn_av_kernel<1><<<ga, 32, 0, stream>>>(ws + OFF_W, ws + OFF_V, ws + OFF_ATT, NH, Nq, Mk, Ms, D);
    gemm(ws + OFF_ATT, bp.o, hbuf, Bc*Nq, D, D, 0, 1);
    ln_kernel<<<Bc*Nq, 32, 0, stream>>>(hbuf, bp.n2.g, bp.n2.b, ws + OFF_X, D);
    gemm(ws + OFF_X, bp.ff.l1, ws + OFF_FFH, Bc*Nq, 4*D, D, 1, 0);
    gemm(ws + OFF_FFH, bp.ff.l2, hbuf, Bc*Nq, D, 4*D, 0, 1);
  };

  // ---- style encoder -> token ----
  mlp2(spos, sposmlp, ws + OFF_HS, Bc*NSn, 3, SHID, SHID, 0);
  add_role_cond_kernel<<<(Bc*NSn*SHID + 255) / 256, 256, 0, stream>>>(
      ws + OFF_HS, sroles, srole_emb, nullptr, Bc*NSn, NSn, SHID);
  ln_kernel<<<Bc*NSn, 32, 0, stream>>>(ws + OFF_HS, s_in.g, s_in.b, ws + OFF_HS, SHID);
  const int smode[3] = {1, 2, 0};  // distance, connectivity, free
  for (int i = 0; i < 3; ++i)
    attn_layer(SL[i], ws + OFF_HS, NSn, SHID, SDH, smode[i], spos, sadj, nullptr, nullptr);
  ln_kernel<<<Bc*NSn, 32, 0, stream>>>(ws + OFF_HS, s_final.g, s_final.b, ws + OFF_X, SHID);
  meanpool_kernel<<<(Bc*SHID + 255) / 256, 256, 0, stream>>>(ws + OFF_X, ws + OFF_TOKIN, NSn, SHID);
  mlp2(ws + OFF_TOKIN, tokproj, ws + OFF_TOK, Bc, SHID, HID, HID, 0);

  // ---- main 6 layers (distance, connectivity, stress, free, distance, connectivity) ----
  const int mmode[4] = {1, 2, 3, 0};
  for (int i = 0; i < 6; ++i)
    attn_layer(L[i], ws + OFF_H, Nn, HID, DHm, mmode[i % 4],
               positions, adjacency, ws + OFF_EHC, ws + OFF_TOK);

  // ---- heads ----
  ln_kernel<<<Bc*Nn, 32, 0, stream>>>(ws + OFF_H, fnorm.g, fnorm.b, ws + OFF_X, HID);
  mlp2(ws + OFF_X, poshead, out + OUT_POS, Bc*Nn, HID, HID, 3, 0);
  mlp2(ws + OFF_X, lathead, out + OUT_LAT, Bc*Nn, HID, HID, LATd, 0);
  dim3 gadj(Nn / 16, Nn / 16, Bc);
  adj_head_kernel<<<gadj, 32, 0, stream>>>(out + OUT_LAT, adjacency, out + OUT_ADJV, out + OUT_PADJ);
}
